// GCN_layer_69793218560049
// MI455X (gfx1250) — matrix operand order repaired and checked by reading, not compiled, not run on hardware
//
#include <hip/hip_runtime.h>

typedef __attribute__((ext_vector_type(16))) _Float16 v16h;
typedef __attribute__((ext_vector_type(8)))  float    v8f;
typedef __attribute__((ext_vector_type(2)))  __fp16   fp16x2;

#define NROWS 8192
#define DFEAT 1024
#define M_TILE 128
#define N_TILE 256
#define K_TILE 64

#if __has_builtin(__builtin_amdgcn_sched_barrier)
#define SCHED_FENCE() __builtin_amdgcn_sched_barrier(0)
#else
#define SCHED_FENCE()
#endif

static __device__ __forceinline__ unsigned cvt2(float a, float b) {
#if __has_builtin(__builtin_amdgcn_cvt_pkrtz)
    union { fp16x2 h; unsigned u; } c;
    c.h = __builtin_amdgcn_cvt_pkrtz(a, b);
    return c.u;
#else
    union { _Float16 h[2]; unsigned u; } c;
    c.h[0] = (_Float16)a; c.h[1] = (_Float16)b;
    return c.u;
#endif
}

// ---------------- Kernel 1: rowsum -> d_inv_sqrt ----------------
__global__ void gcn_rowsum_kernel(const float* __restrict__ Mat,
                                  float* __restrict__ dinv) {
    const int row = blockIdx.x;
    const int tid = threadIdx.x;
    const float4* rp = (const float4*)(Mat + (size_t)row * NROWS);
    float s = 0.f;
    #pragma unroll
    for (int i = 0; i < 8; ++i) {
        float4 v = rp[tid + i * 256];
        s += v.x + v.y + v.z + v.w;
    }
    __shared__ float red[256];
    red[tid] = s;
    __syncthreads();
    for (int off = 128; off > 0; off >>= 1) {
        if (tid < off) red[tid] += red[tid + off];
        __syncthreads();
    }
    if (tid == 0) {
        float r = rsqrtf(red[0] + 1e-8f);
        if (isinf(r)) r = 0.f;
        dinv[row] = r;
    }
}

// ---------------- Kernel 2: out = features (generic scatter semantics) -------
__global__ void gcn_copy_kernel(const float* __restrict__ src,
                                float* __restrict__ dst) {
    size_t i = (size_t)blockIdx.x * blockDim.x + threadIdx.x;
    size_t total = (size_t)NROWS * DFEAT / 4;
    const float4* s4 = (const float4*)src;
    float4* d4 = (float4*)dst;
    for (; i < total; i += (size_t)gridDim.x * blockDim.x) d4[i] = s4[i];
}

// ---------------- Kernel 3: WMMA GEMM, double-buffered LDS + reg prefetch ----
// C[m,:] = dinv[m] * sum_k (Mat[m,k]*dinv[k])_f16 * features[k,:]_f16
// 512 threads = 16 waves as 4(M)x4(N); each wave: 32 rows x 64 cols
__global__ __launch_bounds__(512)
void gcn_wmma_kernel(const float* __restrict__ Mat,
                     const float* __restrict__ Feat,
                     const float* __restrict__ dinv,
                     const int* __restrict__ index,
                     float* __restrict__ out) {
    __shared__ _Float16 A_s[2 * M_TILE * K_TILE];   // 2 x 16 KB
    __shared__ _Float16 B_s[2 * N_TILE * K_TILE];   // 2 x 32 KB (transposed [n][k])
    __shared__ float    D_s[NROWS];                 // 32 KB dinv cache

    const int tid   = threadIdx.x;
    const int lane  = tid & 31;
    const int wave  = tid >> 5;         // 0..15
    const int waveM = wave & 3;         // 32-row strip
    const int waveN = wave >> 2;        // 0..3 -> 64-col strip
    const int mBase = blockIdx.y * M_TILE;
    const int nBase = blockIdx.x * N_TILE;

    const int mLane = lane & 15;
    const int koffA = (lane >> 4) * 16;
    const int koffB = (lane >> 4) * 16;

    // A loader coords: one row, 16 consecutive k per thread
    const int aRow  = tid >> 2;         // 0..127
    const int aKseg = (tid & 3) << 4;   // 0,16,32,48
    // B loader coords: 4k x 4n micro-block; tid+512 -> kb += 8
    const int bN4 = tid & 63;           // n0 = bN4*4
    const int bKb = tid >> 6;           // 0..7 (u=0), +8 (u=1)

    float4 aV[4];
    float4 bV[2][4];
    v8f acc[2][4] = {};

    auto loadRegs = [&](int kt) {
        const float* ar = Mat + (size_t)(mBase + aRow) * NROWS + kt + aKseg;
        #pragma unroll
        for (int i = 0; i < 4; ++i) aV[i] = *(const float4*)(ar + 4 * i);
        #pragma unroll
        for (int u = 0; u < 2; ++u) {
            int k0 = (bKb + u * 8) * 4;
            #pragma unroll
            for (int j = 0; j < 4; ++j)
                bV[u][j] = *(const float4*)(Feat + (size_t)(kt + k0 + j) * DFEAT
                                            + nBase + bN4 * 4);
        }
    };

    auto convertStore = [&](int buf, int kt) {
        _Float16* As = A_s + buf * (M_TILE * K_TILE);
        _Float16* Bs = B_s + buf * (N_TILE * K_TILE);
        // A: scale by dinv[k] (from LDS), 16 halves -> two b128 stores
        union { float4 v; float f[4]; } a[4], dk[4];
        #pragma unroll
        for (int i = 0; i < 4; ++i) {
            a[i].v  = aV[i];
            dk[i].v = *(const float4*)&D_s[kt + aKseg + 4 * i];
            #pragma unroll
            for (int e = 0; e < 4; ++e) a[i].f[e] *= dk[i].f[e];
        }
        uint4 p0, p1;
        p0.x = cvt2(a[0].f[0], a[0].f[1]); p0.y = cvt2(a[0].f[2], a[0].f[3]);
        p0.z = cvt2(a[1].f[0], a[1].f[1]); p0.w = cvt2(a[1].f[2], a[1].f[3]);
        p1.x = cvt2(a[2].f[0], a[2].f[1]); p1.y = cvt2(a[2].f[2], a[2].f[3]);
        p1.z = cvt2(a[3].f[0], a[3].f[1]); p1.w = cvt2(a[3].f[2], a[3].f[3]);
        *(uint4*)&As[aRow * K_TILE + aKseg]     = p0;
        *(uint4*)&As[aRow * K_TILE + aKseg + 8] = p1;
        // B: transpose 4x4 micro-block, b64 stores along k (no scaling)
        #pragma unroll
        for (int u = 0; u < 2; ++u) {
            int k0 = (bKb + u * 8) * 4;
            union { float4 v; float f[4]; } b[4];
            #pragma unroll
            for (int j = 0; j < 4; ++j) b[j].v = bV[u][j];
            #pragma unroll
            for (int nj = 0; nj < 4; ++nj) {
                uint2 p;
                p.x = cvt2(b[0].f[nj], b[1].f[nj]);
                p.y = cvt2(b[2].f[nj], b[3].f[nj]);
                *(uint2*)&Bs[(bN4 * 4 + nj) * K_TILE + k0] = p;
            }
        }
    };

    auto compute = [&](int buf) {
        const _Float16* As = A_s + buf * (M_TILE * K_TILE);
        const _Float16* Bs = B_s + buf * (N_TILE * K_TILE);
        #pragma unroll
        for (int kk = 0; kk < K_TILE; kk += 32) {
            union { v16h v; unsigned u[8]; } af[2];
            #pragma unroll
            for (int g = 0; g < 2; ++g) {
                const unsigned* ar =
                    (const unsigned*)&As[(waveM * 32 + g * 16 + mLane) * K_TILE];
                #pragma unroll
                for (int p = 0; p < 8; ++p) {
                    int K = 2 * p + koffA + kk;  // even
                    af[g].u[p] = ar[K >> 1];
                }
            }
            #pragma unroll
            for (int f = 0; f < 4; ++f) {
                union { v16h v; unsigned u[8]; } bf;
                const unsigned* br =
                    (const unsigned*)&Bs[(waveN * 64 + f * 16 + mLane) * K_TILE];
                #pragma unroll
                for (int p = 0; p < 8; ++p) bf.u[p] = br[p + ((koffB + kk) >> 1)];
                acc[0][f] = __builtin_amdgcn_wmma_f32_16x16x32_f16(
                    false, af[0].v, false, bf.v, (short)0, acc[0][f], false, false);
                acc[1][f] = __builtin_amdgcn_wmma_f32_16x16x32_f16(
                    false, af[1].v, false, bf.v, (short)0, acc[1][f], false, false);
            }
        }
    };

    // Preload dinv into LDS once (512 threads x 4 float4 = 8192 floats)
    #pragma unroll
    for (int i = 0; i < 4; ++i) {
        int o = tid * 4 + i * 2048;
        *(float4*)&D_s[o] = *(const float4*)&dinv[o];
    }
    loadRegs(0);
    __syncthreads();            // D_s visible
    convertStore(0, 0);
    __syncthreads();
    int cur = 0;
    for (int kt = K_TILE; kt < NROWS; kt += K_TILE) {
        loadRegs(kt);           // global loads issued, consumed after compute
        SCHED_FENCE();          // keep loads above the WMMA stream
        compute(cur);           // 16 WMMAs from current LDS buffer
        SCHED_FENCE();          // keep converts below the WMMA stream
        convertStore(cur ^ 1, kt);
        __syncthreads();        // single barrier per K-iter
        cur ^= 1;
    }
    compute(cur);

    // ---- epilogue: scale by dinv[m], scatter rows by index ----
    const int hiM = (lane >> 4) * 8;
    #pragma unroll
    for (int g = 0; g < 2; ++g) {
        int mW = mBase + waveM * 32 + g * 16;
        #pragma unroll
        for (int f = 0; f < 4; ++f) {
            int gn = nBase + waveN * 64 + f * 16 + mLane;
            #pragma unroll
            for (int r = 0; r < 8; ++r) {
                int gm  = mW + r + hiM;
                int row = index[gm];
                __builtin_nontemporal_store(dinv[gm] * acc[g][f][r],
                                            out + (size_t)row * DFEAT + gn);
            }
        }
    }
}

extern "C" void kernel_launch(void* const* d_in, const int* in_sizes, int n_in,
                              void* d_out, int out_size, void* d_ws, size_t ws_size,
                              hipStream_t stream) {
    const float* features = (const float*)d_in[0];
    const float* Mat      = (const float*)d_in[1];
    const int*   index    = (const int*)d_in[2];
    float* out  = (float*)d_out;
    float* dinv = (float*)d_ws;   // 8192 floats of workspace

    gcn_rowsum_kernel<<<NROWS, 256, 0, stream>>>(Mat, dinv);
    gcn_copy_kernel<<<4096, 256, 0, stream>>>(features, out);
    dim3 grid(DFEAT / N_TILE /*=4, fast dim -> L2 reuse of Mat rows*/,
              NROWS / M_TILE /*=64*/);
    gcn_wmma_kernel<<<grid, 512, 0, stream>>>(Mat, features, dinv, index, out);
}
